// TransferNet_22488448761952
// MI455X (gfx1250) — compile-verified, matrix operand order checked by reading, not run on hardware
//
#include <hip/hip_runtime.h>

// ---------------------------------------------------------------------------
// Problem constants (from the reference)
// ---------------------------------------------------------------------------
#define BB   16
#define LQ   32
#define LD   16
#define VV   30000
#define EE   40000
#define TT   10000
#define DW   256
#define HH   512
#define SS   3
#define HD   256
#define G3   (3 * HD)   // 768 gate width

// ---------------------------------------------------------------------------
// CDNA5 WMMA types
// ---------------------------------------------------------------------------
typedef __attribute__((ext_vector_type(16))) __bf16         v16bf;
typedef __attribute__((ext_vector_type(8)))  float          v8f;
typedef __attribute__((ext_vector_type(8)))  unsigned short v8us;

union FragBF { v16bf bf; v8us us[2]; };

__device__ __forceinline__ unsigned short f2bf(float f) {
  unsigned int u = __float_as_uint(f);
  u += 0x7fffu + ((u >> 16) & 1u);   // round-to-nearest-even
  return (unsigned short)(u >> 16);
}
__device__ __forceinline__ unsigned int pack2bf(float lo, float hi) {
  return ((unsigned int)f2bf(hi) << 16) | (unsigned int)f2bf(lo);
}
__device__ __forceinline__ float sigf(float x) {
  return 1.0f / (1.0f + __expf(-x));
}

// ---------------------------------------------------------------------------
// Generic bf16-WMMA GEMM:  C[m,n] = act( bias[n] + sum_k A[m,k] * B[n,k] )
//   A rows optionally gathered: row = Aidx[m*aidx_stride + aidx_off]
//   act: 0 = none, 1 = tanh, 2 = sigmoid ; bias_scalar: bias[0] broadcast
//
// Block = 128 threads (4 waves). Tile = 128(M) x 64(N), K chunked by 32.
// Wave w owns rows w*32..w*32+31 (2 A fragments) x all 64 columns
// (4 B fragments) -> 8 WMMAs per K-chunk with B-fragment reuse.
// REQUIRES: K % 32 == 0 (true for all call sites: K = 256 or 512),
//           A/B rows 16-byte aligned (lda/ldb multiples of 4).
// Out-of-range staging rows are CLAMPED (not zero-filled): the garbage
// tiles they produce are never stored (store side guards m<M, n<N).
// ---------------------------------------------------------------------------
__global__ __launch_bounds__(128)
void gemm_wmma_kernel(const float* __restrict__ A,
                      const int*   __restrict__ Aidx, int aidx_stride, int aidx_off,
                      int lda,
                      const float* __restrict__ Bm, int ldb,
                      const float* __restrict__ bias, int bias_scalar,
                      float* __restrict__ C, int ldc,
                      int M, int N, int K, int act)
{
  __shared__ __align__(16) unsigned short sA[128][32];
  __shared__ __align__(16) unsigned short sB[64][32];

  const int tid   = threadIdx.x;
  const int wave  = tid >> 5;
  const int lane  = tid & 31;
  const int tileM = blockIdx.y * 128;
  const int tileN = blockIdx.x * 64;
  const int hi    = lane >> 4;     // half-wave select (K group)
  const int l15   = lane & 15;

  // --- loop-invariant staging pointers (gather hoisted out of K loop) ---
  int am = tileM + tid;            // one A row per thread
  if (am > M - 1) am = M - 1;      // clamp: garbage never stored
  const long long arowi = Aidx
      ? (long long)Aidx[(long long)am * aidx_stride + aidx_off]
      : (long long)am;
  const float* arow = A + arowi * (long long)lda;

  int bn = tileN + (tid >> 1);     // two threads per B row
  if (bn > N - 1) bn = N - 1;
  const float* brow = Bm + (long long)bn * ldb + (tid & 1) * 16;
  unsigned short* sBdst = &sB[tid >> 1][(tid & 1) * 16];

  // accumulators, initialized with bias (bias depends only on column n)
  v8f acc[2][4];
#pragma unroll
  for (int t = 0; t < 4; ++t) {
    const int n = tileN + t * 16 + l15;
    float bv = 0.0f;
    if (n < N) bv = bias_scalar ? bias[0] : bias[n];
#pragma unroll
    for (int r = 0; r < 8; ++r) { acc[0][t][r] = bv; acc[1][t][r] = bv; }
  }

  for (int k0 = 0; k0 < K; k0 += 32) {
    if (k0 + 32 < K) {                       // prefetch next chunk
      __builtin_prefetch(arow + k0 + 32, 0, 0);
      __builtin_prefetch(brow + k0 + 32, 0, 0);
    }
    __syncthreads();
    // ---- stage A: one 32-float row slice per thread, branch-free ----
    {
      const float4* src = (const float4*)(arow + k0);
      union { unsigned int u[16]; uint4 q[4]; } pk;
#pragma unroll
      for (int q = 0; q < 8; ++q) {
        const float4 v = src[q];
        pk.u[2 * q]     = pack2bf(v.x, v.y);
        pk.u[2 * q + 1] = pack2bf(v.z, v.w);
      }
      uint4* dst = (uint4*)&sA[tid][0];
#pragma unroll
      for (int q = 0; q < 4; ++q) dst[q] = pk.q[q];
    }
    // ---- stage B: one 16-float half-row per thread ----
    {
      const float4* src = (const float4*)(brow + k0);
      union { unsigned int u[8]; uint4 q[2]; } pk;
#pragma unroll
      for (int q = 0; q < 4; ++q) {
        const float4 v = src[q];
        pk.u[2 * q]     = pack2bf(v.x, v.y);
        pk.u[2 * q + 1] = pack2bf(v.z, v.w);
      }
      uint4* dst = (uint4*)sBdst;
      dst[0] = pk.q[0];
      dst[1] = pk.q[1];
    }
    __syncthreads();

    // ---- fragments (ISA 16-bit A layout: lane-half hi selects K groups:
    //      elems 0..7 -> K=hi*8+0..7 ; elems 8..15 -> K=16+hi*8+0..7) ----
    const int mr0 = (wave << 5) + l15;
    const int mr1 = mr0 + 16;
    FragBF fa0, fa1;
    fa0.us[0] = *(const v8us*)&sA[mr0][hi * 8];
    fa0.us[1] = *(const v8us*)&sA[mr0][16 + hi * 8];
    fa1.us[0] = *(const v8us*)&sA[mr1][hi * 8];
    fa1.us[1] = *(const v8us*)&sA[mr1][16 + hi * 8];

#pragma unroll
    for (int t = 0; t < 4; ++t) {
      const int nc = t * 16 + l15;
      FragBF fb;
      fb.us[0] = *(const v8us*)&sB[nc][hi * 8];
      fb.us[1] = *(const v8us*)&sB[nc][16 + hi * 8];
      acc[0][t] = __builtin_amdgcn_wmma_f32_16x16x32_bf16(
          false, fa0.bf, false, fb.bf, (short)0, acc[0][t], false, false);
      acc[1][t] = __builtin_amdgcn_wmma_f32_16x16x32_bf16(
          false, fa1.bf, false, fb.bf, (short)0, acc[1][t], false, false);
    }
  }

  // store: VGPR r of lane holds (m = a*16 + hi*8 + r, n = t*16 + (lane&15))
#pragma unroll
  for (int a = 0; a < 2; ++a) {
#pragma unroll
    for (int t = 0; t < 4; ++t) {
      const int n = tileN + t * 16 + l15;
      if (n >= N) continue;
#pragma unroll
      for (int r = 0; r < 8; ++r) {
        const int m = tileM + (wave << 5) + a * 16 + hi * 8 + r;
        if (m < M) {
          float v = acc[a][t][r];
          if (act == 1)      v = tanhf(v);
          else if (act == 2) v = sigf(v);
          C[(long long)m * ldc + n] = v;
        }
      }
    }
  }
}

// ---------------------------------------------------------------------------
// GRU pointwise step (HD = 256 fixed).
// ---------------------------------------------------------------------------
__global__ void gru_point_kernel(const float* __restrict__ gx, long long gxs,
                                 const float* __restrict__ gh, long long ghs,
                                 const int*   __restrict__ tok, int ts, int toff,
                                 float* __restrict__ h, long long hs_,
                                 float* __restrict__ hseq, long long seqs,
                                 int rows)
{
  long long i = (long long)blockIdx.x * blockDim.x + threadIdx.x;
  if (i >= (long long)rows * HD) return;
  const int n = (int)(i >> 8);
  const int j = (int)(i & 255);
  const float* gxr = gx + (long long)n * gxs;
  const float* ghr = gh + (long long)n * ghs;
  float hv = h[(long long)n * hs_ + j];
  const bool m = tok[(long long)n * ts + toff] != 0;
  const float r  = sigf(gxr[j]        + ghr[j]);
  const float z  = sigf(gxr[HD + j]   + ghr[HD + j]);
  const float nn = tanhf(gxr[2*HD + j] + r * ghr[2*HD + j]);
  float hn = (1.0f - z) * nn + z * hv;
  float out = m ? hn : hv;
  h[(long long)n * hs_ + j] = out;
  if (hseq) hseq[(long long)n * seqs + j] = out;
}

__global__ void zero_kernel(float* __restrict__ p, long long n) {
  long long i = (long long)blockIdx.x * blockDim.x + threadIdx.x;
  if (i < n) p[i] = 0.0f;
}

__global__ void copy_rows_kernel(const float* __restrict__ src, long long ss,
                                 float* __restrict__ dst, long long ds_,
                                 int rows, int cols) {
  long long i = (long long)blockIdx.x * blockDim.x + threadIdx.x;
  if (i >= (long long)rows * cols) return;
  int r = (int)(i / cols), c = (int)(i % cols);
  dst[(long long)r * ds_ + c] = src[(long long)r * ss + c];
}

// ---------------------------------------------------------------------------
// Attention over question words + context, fused with *rel_w scaling.
// ---------------------------------------------------------------------------
__global__ void attn_ctx_kernel(const float* __restrict__ cq,
                                const float* __restrict__ qwh,   // [B,LQ,H]
                                const int*   __restrict__ questions,
                                const float* __restrict__ rel_w,
                                float* __restrict__ ctx_scaled)  // [B,H]
{
  const int b = blockIdx.x;
  __shared__ float slog[LQ];
  __shared__ float sdist[LQ];

  const int l   = threadIdx.x >> 3;
  const int sub = threadIdx.x & 7;
  float partial = 0.0f;
  for (int j = sub; j < HH; j += 8)
    partial += cq[b * HH + j] * qwh[((long long)b * LQ + l) * HH + j];
  for (int o = 4; o; o >>= 1) partial += __shfl_down(partial, o, 8);
  if (sub == 0) slog[l] = partial;
  __syncthreads();

  if (threadIdx.x < LQ) {
    float mx = -1e30f;
    for (int i = 0; i < LQ; ++i) mx = fmaxf(mx, slog[i]);
    float s = 0.0f;
    for (int i = 0; i < LQ; ++i) s += __expf(slog[i] - mx);
    float d = __expf(slog[threadIdx.x] - mx) / s;
    if (questions[b * LQ + threadIdx.x] == 0) d = 0.0f;   // mask
    sdist[threadIdx.x] = d;
  }
  __syncthreads();

  float dsum = 0.0f;
  for (int i = 0; i < LQ; ++i) dsum = dsum + sdist[i];
  const float inv = 1.0f / (dsum + 1e-6f);

  for (int j = threadIdx.x; j < HH; j += 256) {
    float a = 0.0f;
    for (int l2 = 0; l2 < LQ; ++l2)
      a += sdist[l2] * qwh[((long long)b * LQ + l2) * HH + j];
    ctx_scaled[b * HH + j] = (a * inv + cq[b * HH + j]) * rel_w[j];
  }
}

// ---------------------------------------------------------------------------
// Sparse "follow": e_new[b, obj[t]] += e_cur[b, sub[t]] * d_prob[b, t]
// ---------------------------------------------------------------------------
__global__ void follow_kernel(const float* __restrict__ e_cur,
                              const float* __restrict__ d_prob,
                              const int*   __restrict__ kb_pair,
                              float* __restrict__ e_new)
{
  long long i = (long long)blockIdx.x * blockDim.x + threadIdx.x;
  if (i >= (long long)BB * TT) return;
  const int b = (int)(i / TT);
  const int t = (int)(i % TT);
  const int s = kb_pair[2 * t];
  const int o = kb_pair[2 * t + 1];
  const float v = e_cur[(long long)b * EE + s] * d_prob[(long long)b * TT + t];
  atomicAdd(&e_new[(long long)b * EE + o], v);
}

__global__ void hop_store_kernel(const float* __restrict__ e_new,
                                 float* __restrict__ e_cur,
                                 float* __restrict__ hop_res, int t)
{
  long long i = (long long)blockIdx.x * blockDim.x + threadIdx.x;
  if (i >= (long long)BB * EE) return;
  const int b = (int)(i / EE);
  const int e = (int)(i % EE);
  float v = e_new[i];
  float z = fmaxf(v, 1.0f);
  v = v / z;
  e_cur[i] = v;
  hop_res[(long long)b * (SS * EE) + (long long)t * EE + e] = v;
}

__global__ void hop_attn_kernel(const float* __restrict__ q_emb,
                                const float* __restrict__ hop_W,
                                const float* __restrict__ hop_b,
                                float* __restrict__ hop_attn)
{
  const int b = threadIdx.x;
  if (b >= BB) return;
  float lg[SS];
  for (int s = 0; s < SS; ++s) {
    float a = hop_b[s];
    for (int j = 0; j < HH; ++j) a += q_emb[b * HH + j] * hop_W[s * HH + j];
    lg[s] = a;
  }
  float mx = fmaxf(lg[0], fmaxf(lg[1], lg[2]));
  float e0 = __expf(lg[0] - mx), e1 = __expf(lg[1] - mx), e2 = __expf(lg[2] - mx);
  const float inv = 1.0f / (e0 + e1 + e2);
  hop_attn[b * SS + 0] = e0 * inv;
  hop_attn[b * SS + 1] = e1 * inv;
  hop_attn[b * SS + 2] = e2 * inv;
}

__global__ void final_kernel(const float* __restrict__ hop_res,
                             const float* __restrict__ hop_attn,
                             const float* __restrict__ e_s,
                             const float* __restrict__ q_mask,
                             float* __restrict__ out)
{
  long long i = (long long)blockIdx.x * blockDim.x + threadIdx.x;
  if (i >= (long long)BB * EE) return;
  const int b = (int)(i / EE);
  const int e = (int)(i % EE);
  const float a0 = hop_attn[b * SS + 0];
  const float a1 = hop_attn[b * SS + 1];
  const float a2 = hop_attn[b * SS + 2];
  const long long base = (long long)b * (SS * EE) + e;
  float v = hop_res[base] * a0 + hop_res[base + EE] * a1 + hop_res[base + 2 * EE] * a2;
  const bool am1 = (a1 > a0) && (a1 >= a2);   // jnp.argmax == 1 (first-max rule)
  const float m = am1 ? e_s[i] : 0.0f;
  v = (1.0f - m) * v;
  out[i] = v * q_mask[i];
}

// ---------------------------------------------------------------------------
// Host orchestration
// ---------------------------------------------------------------------------
extern "C" void kernel_launch(void* const* d_in, const int* in_sizes, int n_in,
                              void* d_out, int out_size, void* d_ws, size_t ws_size,
                              hipStream_t stream) {
  (void)in_sizes; (void)n_in; (void)out_size; (void)ws_size;

  const int*   questions = (const int*)  d_in[0];
  const float* e_s       = (const float*)d_in[1];
  const int*   kb_pair   = (const int*)  d_in[2];
  const int*   kb_desc   = (const int*)  d_in[3];
  const float* word_emb  = (const float*)d_in[4];
  const float* step_W    = (const float*)d_in[5];
  const float* step_b    = (const float*)d_in[6];
  const float* rel_w     = (const float*)d_in[7];
  const float* rel_b     = (const float*)d_in[8];
  const float* qcls_W    = (const float*)d_in[9];
  const float* qcls_b    = (const float*)d_in[10];
  const float* hop_W     = (const float*)d_in[11];
  const float* hop_b     = (const float*)d_in[12];
  const float* qf_W = (const float*)d_in[13], *qf_U = (const float*)d_in[14],
             * qf_bi = (const float*)d_in[15], *qf_bh = (const float*)d_in[16];
  const float* qb_W = (const float*)d_in[17], *qb_U = (const float*)d_in[18],
             * qb_bi = (const float*)d_in[19], *qb_bh = (const float*)d_in[20];
  const float* df_W = (const float*)d_in[21], *df_U = (const float*)d_in[22],
             * df_bi = (const float*)d_in[23], *df_bh = (const float*)d_in[24];
  const float* db_W = (const float*)d_in[25], *db_U = (const float*)d_in[26],
             * db_bi = (const float*)d_in[27], *db_bh = (const float*)d_in[28];

  // ----- workspace partition (floats) -----
  float* ws = (float*)d_ws;
  size_t off = 0;
  auto alloc = [&](size_t n) { float* p = ws + off; off += n; return p; };
  float* q_xw_f   = alloc((size_t)BB * LQ * G3);
  float* q_xw_b   = alloc((size_t)BB * LQ * G3);
  float* q_gh     = alloc((size_t)BB * G3);
  float* q_h      = alloc((size_t)BB * HD);
  float* q_word_h = alloc((size_t)BB * LQ * HH);
  float* q_emb    = alloc((size_t)BB * HH);
  float* d_gx     = alloc((size_t)TT * G3);
  float* d_gh     = alloc((size_t)TT * G3);
  float* d_h      = alloc((size_t)TT * HD);
  float* desc_emb = alloc((size_t)TT * HH);
  float* cq       = alloc((size_t)BB * HH);
  float* ctxs     = alloc((size_t)BB * HH);
  float* d_prob   = alloc((size_t)BB * TT);
  float* e_cur    = alloc((size_t)BB * EE);
  float* e_new    = alloc((size_t)BB * EE);
  float* hop_res  = alloc((size_t)BB * SS * EE);
  float* hop_att  = alloc((size_t)BB * SS);
  float* q_maskb  = alloc((size_t)BB * EE);

  const dim3 blk128(128);
  auto gemm = [&](const float* A, const int* Aidx, int aist, int aioff, int lda,
                  const float* Bm, int ldb, const float* bias, int bsc,
                  float* C, int ldc, int M, int N, int K, int act) {
    dim3 grid((N + 63) / 64, (M + 127) / 128);
    gemm_wmma_kernel<<<grid, blk128, 0, stream>>>(A, Aidx, aist, aioff, lda,
                                                  Bm, ldb, bias, bsc,
                                                  C, ldc, M, N, K, act);
  };
  auto zero = [&](float* p, long long n) {
    zero_kernel<<<(unsigned)((n + 255) / 256), 256, 0, stream>>>(p, n);
  };
  auto copy_rows = [&](const float* src, long long ss, float* dst, long long ds_,
                       int rows, int cols) {
    long long n = (long long)rows * cols;
    copy_rows_kernel<<<(unsigned)((n + 255) / 256), 256, 0, stream>>>(
        src, ss, dst, ds_, rows, cols);
  };
  auto gru_point = [&](const float* gx, long long gxs, const float* gh, long long ghs,
                       const int* tok, int ts, int toff,
                       float* h, float* hseq, long long seqs, int rows) {
    long long n = (long long)rows * HD;
    gru_point_kernel<<<(unsigned)((n + 255) / 256), 256, 0, stream>>>(
        gx, gxs, gh, ghs, tok, ts, toff, h, (long long)HD, hseq, seqs, rows);
  };

  // ===== Question BiGRU =====
  gemm(word_emb, questions, 1, 0, DW, qf_W, DW, qf_bi, 0, q_xw_f, G3,
       BB * LQ, G3, DW, 0);
  gemm(word_emb, questions, 1, 0, DW, qb_W, DW, qb_bi, 0, q_xw_b, G3,
       BB * LQ, G3, DW, 0);

  zero(q_h, (long long)BB * HD);
  for (int l = 0; l < LQ; ++l) {
    gemm(q_h, nullptr, 0, 0, HD, qf_U, HD, qf_bh, 0, q_gh, G3, BB, G3, HD, 0);
    gru_point(q_xw_f + (long long)l * G3, (long long)LQ * G3, q_gh, G3,
              questions, LQ, l, q_h, q_word_h + (long long)l * HH,
              (long long)LQ * HH, BB);
  }
  copy_rows(q_h, HD, q_emb, HH, BB, HD);

  zero(q_h, (long long)BB * HD);
  for (int l = LQ - 1; l >= 0; --l) {
    gemm(q_h, nullptr, 0, 0, HD, qb_U, HD, qb_bh, 0, q_gh, G3, BB, G3, HD, 0);
    gru_point(q_xw_b + (long long)l * G3, (long long)LQ * G3, q_gh, G3,
              questions, LQ, l, q_h, q_word_h + (long long)l * HH + HD,
              (long long)LQ * HH, BB);
  }
  copy_rows(q_h, HD, q_emb + HD, HH, BB, HD);

  // ===== Description BiGRU (the FLOP-dominant part) =====
  zero(d_h, (long long)TT * HD);
  for (int l = 0; l < LD; ++l) {
    gemm(word_emb, kb_desc, LD, l, DW, df_W, DW, df_bi, 0, d_gx, G3, TT, G3, DW, 0);
    gemm(d_h, nullptr, 0, 0, HD, df_U, HD, df_bh, 0, d_gh, G3, TT, G3, HD, 0);
    gru_point(d_gx, G3, d_gh, G3, kb_desc, LD, l, d_h, nullptr, 0, TT);
  }
  copy_rows(d_h, HD, desc_emb, HH, TT, HD);

  zero(d_h, (long long)TT * HD);
  for (int l = LD - 1; l >= 0; --l) {
    gemm(word_emb, kb_desc, LD, l, DW, db_W, DW, db_bi, 0, d_gx, G3, TT, G3, DW, 0);
    gemm(d_h, nullptr, 0, 0, HD, db_U, HD, db_bh, 0, d_gh, G3, TT, G3, HD, 0);
    gru_point(d_gx, G3, d_gh, G3, kb_desc, LD, l, d_h, nullptr, 0, TT);
  }
  copy_rows(d_h, HD, desc_emb + HD, HH, TT, HD);

  // ===== Hop loop =====
  hipMemcpyAsync(e_cur, e_s, (size_t)BB * EE * sizeof(float),
                 hipMemcpyDeviceToDevice, stream);
  hop_attn_kernel<<<1, 32, 0, stream>>>(q_emb, hop_W, hop_b, hop_att);

  for (int t = 0; t < SS; ++t) {
    gemm(q_emb, nullptr, 0, 0, HH, step_W + (long long)t * HH * HH, HH,
         step_b + (long long)t * HH, 0, cq, HH, BB, HH, HH, /*tanh*/1);
    attn_ctx_kernel<<<BB, 256, 0, stream>>>(cq, q_word_h, questions, rel_w, ctxs);
    gemm(ctxs, nullptr, 0, 0, HH, desc_emb, HH, rel_b, /*scalar*/1,
         d_prob, TT, BB, TT, HH, /*sigmoid*/2);
    zero(e_new, (long long)BB * EE);
    follow_kernel<<<(BB * TT + 255) / 256, 256, 0, stream>>>(e_cur, d_prob,
                                                             kb_pair, e_new);
    hop_store_kernel<<<(BB * EE + 255) / 256, 256, 0, stream>>>(e_new, e_cur,
                                                                hop_res, t);
  }

  // ===== Entity classifier mask + final combine =====
  gemm(q_emb, nullptr, 0, 0, HH, qcls_W, HH, qcls_b, 0, q_maskb, EE,
       BB, EE, HH, /*sigmoid*/2);
  final_kernel<<<(BB * EE + 255) / 256, 256, 0, stream>>>(
      hop_res, hop_att, e_s, q_maskb, (float*)d_out);
}
